// FlashMLAttentionTPU_5068061409580
// MI455X (gfx1250) — matrix operand order repaired and checked by reading, not compiled, not run on hardware
//
#include <hip/hip_runtime.h>
#include <hip/hip_bf16.h>

// ---------------------------------------------------------------------------
// MHA forward for B=1, S=2048, E=1024, H=16, D=64, causal, sm_scale=0.5
// All matmuls via v_wmma_f32_16x16x32_bf16 (CDNA5 wave32 WMMA).
// GEMM: 32x64 per-wave tile; explicit SSA ping-pong double buffering
//       (no runtime-indexed fragment arrays -> no alloca/scratch spills).
// Attention: flash loop with global_prefetch_b8 look-ahead on K/V tiles.
// ---------------------------------------------------------------------------

typedef __bf16 bf16_t;
typedef __attribute__((ext_vector_type(16))) __bf16 v16bf;
typedef __attribute__((ext_vector_type(8)))  __bf16 v8bf;
typedef __attribute__((ext_vector_type(8)))  float  v8f;

#define S_LEN 2048
#define EMB   1024
#define HEADS 16
#define HD    64
#define SM_SCALE 0.5f

union AFrag { v16bf v; v8bf h[2]; };

__device__ inline v8f wmma_bf16f32(const AFrag& a, const AFrag& b, v8f c) {
  // D = A(16x32 bf16) * B(32x16 bf16) + C(16x16 f32)
  return __builtin_amdgcn_wmma_f32_16x16x32_bf16(
      /*neg_a=*/false, a.v, /*neg_b=*/false, b.v,
      /*c_mod=*/(short)0, c, /*reuse_a=*/false, /*reuse_b=*/false);
}

__device__ inline v8f vzero8() {
  v8f z = {0.f, 0.f, 0.f, 0.f, 0.f, 0.f, 0.f, 0.f};
  return z;
}

// A fragment (16x32 bf16): lane = row, elem i -> K = half*16 + i
// (contiguous 32B per lane, matches B fragment order).
__device__ inline void load_afrag(AFrag& f, const bf16_t* row, int kk, int half) {
  f.h[0] = *(const v8bf*)(row + kk + half * 16);
  f.h[1] = *(const v8bf*)(row + kk + half * 16 + 8);
}

// B fragment (32x16 bf16, contiguous k order): lane = col, elem i -> K = half*16 + i.
// `browK` pre-offset by half*16; contiguous 32B per lane.
__device__ inline void load_bfrag(AFrag& f, const bf16_t* browK, int kk) {
  f.h[0] = *(const v8bf*)(browK + kk);
  f.h[1] = *(const v8bf*)(browK + kk + 8);
}

// ---------------------------------------------------------------------------
// Elementwise conversion kernels
// ---------------------------------------------------------------------------
__global__ void f32_to_bf16_kernel(const float* __restrict__ in,
                                   bf16_t* __restrict__ out, int n) {
  int i = blockIdx.x * blockDim.x + threadIdx.x;
  if (i < n) out[i] = (bf16_t)in[i];
}

// out[c*rows + r] = (bf16) in[r*cols + c]   (store W^T so GEMM B-frags are
// contiguous in the K dimension per lane)
__global__ void transpose_to_bf16_kernel(const float* __restrict__ in,
                                         bf16_t* __restrict__ out,
                                         int rows, int cols) {
  int i = blockIdx.x * blockDim.x + threadIdx.x;
  if (i < rows * cols) {
    int r = i / cols, c = i % cols;
    out[(size_t)c * rows + r] = (bf16_t)in[i];
  }
}

// ---------------------------------------------------------------------------
// GEMM:  out[M,N] = A[M,K](bf16, row-major) * BT[N,K](bf16)^T + bias
// One wave computes a 32x64 tile (2 m-subtiles x 4 n-subtiles = 8 WMMA accs).
// K-loop step 64 with explicit ping/pong fragment sets (all constant-indexed,
// so everything stays in VGPRs). Requires K % 64 == 0 (K = 1024 here).
// mode 0: fp32 output [M,N]
// mode 1: bf16 head-split output [H][M][64]  (col -> h = col/64, d = col%64)
// ---------------------------------------------------------------------------
__global__ __launch_bounds__(256, 1) void gemm_bf16_wmma(
    const bf16_t* __restrict__ A, const bf16_t* __restrict__ BT,
    const float* __restrict__ bias, float* __restrict__ outF,
    bf16_t* __restrict__ outH, int M, int N, int K, int mode) {
  const int lane = threadIdx.x & 31;
  const int wave = threadIdx.x >> 5;
  const int ngrp_count = N >> 6;
  const int task = blockIdx.x * 8 + wave;
  if (task >= (M >> 5) * ngrp_count) return;
  const int mtile = task / ngrp_count;
  const int ngrp  = task % ngrp_count;
  const int m0 = mtile * 32, n0 = ngrp * 64;
  const int half = lane >> 4, ln = lane & 15;

  v8f acc[2][4];
#pragma unroll
  for (int mt = 0; mt < 2; ++mt)
#pragma unroll
    for (int nt = 0; nt < 4; ++nt) acc[mt][nt] = vzero8();

  const bf16_t* arow0 = A + (size_t)(m0 + ln) * K;
  const bf16_t* arow1 = A + (size_t)(m0 + 16 + ln) * K;
  const bf16_t* brow0 = BT + (size_t)(n0 + ln) * K + half * 16;
  const bf16_t* brow1 = BT + (size_t)(n0 + 16 + ln) * K + half * 16;
  const bf16_t* brow2 = BT + (size_t)(n0 + 32 + ln) * K + half * 16;
  const bf16_t* brow3 = BT + (size_t)(n0 + 48 + ln) * K + half * 16;

  // Explicit SSA ping (x) / pong (y) fragment sets — no runtime indexing.
  AFrag a0x, a1x, bx0, bx1, bx2, bx3;
  AFrag a0y, a1y, by0, by1, by2, by3;

  load_afrag(a0x, arow0, 0, half);
  load_afrag(a1x, arow1, 0, half);
  load_bfrag(bx0, brow0, 0);
  load_bfrag(bx1, brow1, 0);
  load_bfrag(bx2, brow2, 0);
  load_bfrag(bx3, brow3, 0);

  for (int kk = 0; kk < K; kk += 64) {
    // pong loads @ kk+32 overlap with ping WMMAs
    load_afrag(a0y, arow0, kk + 32, half);
    load_afrag(a1y, arow1, kk + 32, half);
    load_bfrag(by0, brow0, kk + 32);
    load_bfrag(by1, brow1, kk + 32);
    load_bfrag(by2, brow2, kk + 32);
    load_bfrag(by3, brow3, kk + 32);

    acc[0][0] = wmma_bf16f32(a0x, bx0, acc[0][0]);
    acc[1][0] = wmma_bf16f32(a1x, bx0, acc[1][0]);
    acc[0][1] = wmma_bf16f32(a0x, bx1, acc[0][1]);
    acc[1][1] = wmma_bf16f32(a1x, bx1, acc[1][1]);
    acc[0][2] = wmma_bf16f32(a0x, bx2, acc[0][2]);
    acc[1][2] = wmma_bf16f32(a1x, bx2, acc[1][2]);
    acc[0][3] = wmma_bf16f32(a0x, bx3, acc[0][3]);
    acc[1][3] = wmma_bf16f32(a1x, bx3, acc[1][3]);

    // ping loads @ kk+64 overlap with pong WMMAs
    if (kk + 64 < K) {
      load_afrag(a0x, arow0, kk + 64, half);
      load_afrag(a1x, arow1, kk + 64, half);
      load_bfrag(bx0, brow0, kk + 64);
      load_bfrag(bx1, brow1, kk + 64);
      load_bfrag(bx2, brow2, kk + 64);
      load_bfrag(bx3, brow3, kk + 64);
    }

    acc[0][0] = wmma_bf16f32(a0y, by0, acc[0][0]);
    acc[1][0] = wmma_bf16f32(a1y, by0, acc[1][0]);
    acc[0][1] = wmma_bf16f32(a0y, by1, acc[0][1]);
    acc[1][1] = wmma_bf16f32(a1y, by1, acc[1][1]);
    acc[0][2] = wmma_bf16f32(a0y, by2, acc[0][2]);
    acc[1][2] = wmma_bf16f32(a1y, by2, acc[1][2]);
    acc[0][3] = wmma_bf16f32(a0y, by3, acc[0][3]);
    acc[1][3] = wmma_bf16f32(a1y, by3, acc[1][3]);
  }

#pragma unroll
  for (int mt = 0; mt < 2; ++mt) {
#pragma unroll
    for (int nt = 0; nt < 4; ++nt) {
#pragma unroll
      for (int r = 0; r < 8; ++r) {
        const int row = m0 + mt * 16 + half * 8 + r;  // C layout: lanes 0-15
        const int col = n0 + nt * 16 + ln;            // rows 0-7; 16-31: 8-15
        const float v = acc[mt][nt][r] + bias[col];
        if (mode == 0) {
          outF[(size_t)row * N + col] = v;
        } else {
          const int hh = col >> 6, d = col & 63;
          outH[((size_t)hh * M + row) * HD + d] = (bf16_t)v;
        }
      }
    }
  }
}

// ---------------------------------------------------------------------------
// Causal flash attention. One wave per (head, 16-query tile).
// Q/K/V: bf16 [H][S][64]. Out: bf16 merged-head [S][E].
// Key tiles of 32; online softmax; P converted C->A layout via per-wave LDS;
// V staged transposed in LDS so P*V B-fragments are contiguous ds loads.
// Same-wave DS ordering makes per-wave LDS reuse safe without block barriers.
// global_prefetch_b8 look-ahead pulls next K/V tile toward the WGP.
// ---------------------------------------------------------------------------
__global__ __launch_bounds__(256, 1) void flash_attn_wmma(
    const bf16_t* __restrict__ Qh, const bf16_t* __restrict__ Kh,
    const bf16_t* __restrict__ Vh, bf16_t* __restrict__ Out) {
  __shared__ __align__(16) char smem[8 * 5120];  // per wave: Vt 4KB + P 1KB
  const int lane = threadIdx.x & 31;
  const int wave = threadIdx.x >> 5;
  const int task = blockIdx.x * 8 + wave;
  if (task >= HEADS * (S_LEN / 16)) return;
  const int h  = task >> 7;            // 128 query tiles per head
  const int q0 = (task & 127) << 4;
  const int half = lane >> 4, ln = lane & 15;

  bf16_t* ldsVt = (bf16_t*)(smem + wave * 5120);         // [64 d][32 key]
  bf16_t* ldsP  = (bf16_t*)(smem + wave * 5120 + 4096);  // [16 q][32 key]

  const bf16_t* Qp = Qh + (size_t)h * S_LEN * HD;
  const bf16_t* Kp = Kh + (size_t)h * S_LEN * HD;
  const bf16_t* Vp = Vh + (size_t)h * S_LEN * HD;

  // Q A-fragments, K-dim = head-dim split as d=[0,32) and [32,64)
  AFrag qa0, qa1;
  {
    const bf16_t* qrow = Qp + (size_t)(q0 + ln) * HD;
    qa0.h[0] = *(const v8bf*)(qrow + half * 16);
    qa0.h[1] = *(const v8bf*)(qrow + half * 16 + 8);
    qa1.h[0] = *(const v8bf*)(qrow + 32 + half * 16);
    qa1.h[1] = *(const v8bf*)(qrow + 32 + half * 16 + 8);
  }

  v8f o[4];
#pragma unroll
  for (int nt = 0; nt < 4; ++nt) o[nt] = vzero8();
  float m[8], l[8];
#pragma unroll
  for (int r = 0; r < 8; ++r) { m[r] = -__builtin_inff(); l[r] = 0.f; }

  const int kend = q0 + 15;
  for (int k0 = 0; k0 <= kend; k0 += 32) {
    // look-ahead: each lane prefetches one distinct 128B K/V row of next tile
    if (k0 + 32 <= kend) {
      __builtin_prefetch(Kp + (size_t)(k0 + 32 + lane) * HD, 0, 0);
      __builtin_prefetch(Vp + (size_t)(k0 + 32 + lane) * HD, 0, 0);
    }

    // --- K^T B-fragments: B[k=d][n=key]; lane reads its key row, d contiguous
    AFrag kb00, kb01, kb10, kb11;  // kb<subtile><dbase>
    {
      const bf16_t* krow0 = Kp + (size_t)(k0 + ln) * HD;
      const bf16_t* krow1 = Kp + (size_t)(k0 + 16 + ln) * HD;
      kb00.h[0] = *(const v8bf*)(krow0 + half * 16);
      kb00.h[1] = *(const v8bf*)(krow0 + half * 16 + 8);
      kb01.h[0] = *(const v8bf*)(krow0 + 32 + half * 16);
      kb01.h[1] = *(const v8bf*)(krow0 + 32 + half * 16 + 8);
      kb10.h[0] = *(const v8bf*)(krow1 + half * 16);
      kb10.h[1] = *(const v8bf*)(krow1 + half * 16 + 8);
      kb11.h[0] = *(const v8bf*)(krow1 + 32 + half * 16);
      kb11.h[1] = *(const v8bf*)(krow1 + 32 + half * 16 + 8);
    }

    // --- stage V tile transposed in LDS: Vt[d][key]
    {
      const bf16_t* vrow = Vp + (size_t)(k0 + lane) * HD;
      v8bf ch[8];
#pragma unroll
      for (int c = 0; c < 8; ++c) ch[c] = *(const v8bf*)(vrow + c * 8);
#pragma unroll
      for (int c = 0; c < 8; ++c)
#pragma unroll
        for (int e = 0; e < 8; ++e)
          ldsVt[(c * 8 + e) * 32 + lane] = ch[c][e];
    }

    // --- scores S = Q * K^T  (two 16-key subtiles, K-dim=64 -> 2 WMMA each)
    v8f s0 = vzero8(), s1 = vzero8();
    s0 = wmma_bf16f32(qa0, kb00, s0);
    s0 = wmma_bf16f32(qa1, kb01, s0);
    s1 = wmma_bf16f32(qa0, kb10, s1);
    s1 = wmma_bf16f32(qa1, kb11, s1);

    // --- scale, causal mask, online softmax (rows live across 16 lanes/half)
#pragma unroll
    for (int r = 0; r < 8; ++r) {
      const int qrow = q0 + half * 8 + r;
      const int c0 = k0 + ln, c1 = k0 + 16 + ln;
      float v0 = s0[r] * SM_SCALE;
      float v1 = s1[r] * SM_SCALE;
      if (c0 > qrow) v0 = -__builtin_inff();
      if (c1 > qrow) v1 = -__builtin_inff();
      float tmax = fmaxf(v0, v1);
#pragma unroll
      for (int off = 1; off < 16; off <<= 1)
        tmax = fmaxf(tmax, __shfl_xor(tmax, off, 32));
      const float newm = fmaxf(m[r], tmax);
      const float sc = __expf(m[r] - newm);
      const float p0 = __expf(v0 - newm);
      const float p1 = __expf(v1 - newm);
      float rs = p0 + p1;
#pragma unroll
      for (int off = 1; off < 16; off <<= 1)
        rs += __shfl_xor(rs, off, 32);
      l[r] = l[r] * sc + rs;
      m[r] = newm;
#pragma unroll
      for (int nt = 0; nt < 4; ++nt) o[nt][r] *= sc;
      // write P (C layout) to LDS as plain [q][key] bf16
      ldsP[(half * 8 + r) * 32 + ln]      = (bf16_t)p0;
      ldsP[(half * 8 + r) * 32 + 16 + ln] = (bf16_t)p1;
    }

    // --- P A-fragment (16 q x 32 key) from LDS
    AFrag pf;
    pf.h[0] = *(const v8bf*)(ldsP + ln * 32 + half * 16);
    pf.h[1] = *(const v8bf*)(ldsP + ln * 32 + half * 16 + 8);

    // --- O += P * V  (B[k=key][n=d] = Vt[d][key] contiguous per lane)
#pragma unroll
    for (int nt = 0; nt < 4; ++nt) {
      const bf16_t* vt = ldsVt + (nt * 16 + ln) * 32 + half * 16;
      AFrag vb;
      vb.h[0] = *(const v8bf*)(vt);
      vb.h[1] = *(const v8bf*)(vt + 8);
      o[nt] = wmma_bf16f32(pf, vb, o[nt]);
    }
  }

  // --- normalize and store merged-head bf16 [S][E]
#pragma unroll
  for (int nt = 0; nt < 4; ++nt) {
#pragma unroll
    for (int r = 0; r < 8; ++r) {
      const int row = q0 + half * 8 + r;
      const int col = h * HD + nt * 16 + ln;
      Out[(size_t)row * EMB + col] = (bf16_t)(o[nt][r] / l[r]);
    }
  }
}

// ---------------------------------------------------------------------------
// Launcher
// ---------------------------------------------------------------------------
extern "C" void kernel_launch(void* const* d_in, const int* in_sizes, int n_in,
                              void* d_out, int out_size, void* d_ws, size_t ws_size,
                              hipStream_t stream) {
  const float* x  = (const float*)d_in[0];
  const float* Wq = (const float*)d_in[1];
  const float* bq = (const float*)d_in[2];
  const float* Wk = (const float*)d_in[3];
  const float* bk = (const float*)d_in[4];
  const float* Wv = (const float*)d_in[5];
  const float* bv = (const float*)d_in[6];
  const float* Wo = (const float*)d_in[7];
  const float* bo = (const float*)d_in[8];
  float* out = (float*)d_out;

  char* ws = (char*)d_ws;
  const size_t MB = 1024 * 1024;
  // layout (24 MB total): xb reused as attn output after QKV GEMMs are done
  bf16_t* xb   = (bf16_t*)(ws);            // 4 MB  [S,E] bf16
  bf16_t* attn = (bf16_t*)(ws);            // reuse: [S,E] bf16 (after GEMMs)
  bf16_t* WqT  = (bf16_t*)(ws + 4 * MB);   // 2 MB each, [N][K] bf16
  bf16_t* WkT  = (bf16_t*)(ws + 6 * MB);
  bf16_t* WvT  = (bf16_t*)(ws + 8 * MB);
  bf16_t* WoT  = (bf16_t*)(ws + 10 * MB);
  bf16_t* Qh   = (bf16_t*)(ws + 12 * MB);  // 4 MB each, [H][S][64] bf16
  bf16_t* Kh   = (bf16_t*)(ws + 16 * MB);
  bf16_t* Vh   = (bf16_t*)(ws + 20 * MB);

  const int nx = S_LEN * EMB;
  f32_to_bf16_kernel<<<(nx + 255) / 256, 256, 0, stream>>>(x, xb, nx);

  const int nw = EMB * EMB;
  transpose_to_bf16_kernel<<<(nw + 255) / 256, 256, 0, stream>>>(Wq, WqT, EMB, EMB);
  transpose_to_bf16_kernel<<<(nw + 255) / 256, 256, 0, stream>>>(Wk, WkT, EMB, EMB);
  transpose_to_bf16_kernel<<<(nw + 255) / 256, 256, 0, stream>>>(Wv, WvT, EMB, EMB);
  transpose_to_bf16_kernel<<<(nw + 255) / 256, 256, 0, stream>>>(Wo, WoT, EMB, EMB);

  // 32x64 per-wave tiles: (2048/32)*(1024/64) = 1024 wave tasks, 8 waves/block
  const int gemm_blocks = (S_LEN / 32) * (EMB / 64) / 8;  // 128
  gemm_bf16_wmma<<<gemm_blocks, 256, 0, stream>>>(xb, WqT, bq, nullptr, Qh,
                                                  S_LEN, EMB, EMB, 1);
  gemm_bf16_wmma<<<gemm_blocks, 256, 0, stream>>>(xb, WkT, bk, nullptr, Kh,
                                                  S_LEN, EMB, EMB, 1);
  gemm_bf16_wmma<<<gemm_blocks, 256, 0, stream>>>(xb, WvT, bv, nullptr, Vh,
                                                  S_LEN, EMB, EMB, 1);

  const int attn_blocks = HEADS * (S_LEN / 16) / 8;  // 256
  flash_attn_wmma<<<attn_blocks, 256, 0, stream>>>(Qh, Kh, Vh, attn);

  gemm_bf16_wmma<<<gemm_blocks, 256, 0, stream>>>(attn, WoT, bo, out, nullptr,
                                                  S_LEN, EMB, EMB, 0);
}
